// GaussianMixtureUnif_basic_62680752718035
// MI455X (gfx1250) — matrix operand order repaired and checked by reading, not compiled, not run on hardware
//
#include <hip/hip_runtime.h>
#include <hip/hip_bf16.h>
#include <cstdint>

// Problem sizes fixed by the reference
constexpr int N = 32768;
constexpr int C = 1024;
constexpr int D = 128;

typedef __attribute__((ext_vector_type(16))) __bf16 v16bf;
typedef __attribute__((ext_vector_type(8)))  __bf16 v8bf;
typedef __attribute__((ext_vector_type(8)))  float  v8f;

__device__ __forceinline__ v8f wmma_bf16(v16bf a, v16bf b, v8f c) {
  // (neg_a, A, neg_b, B, c_mod, C, reuse_a, reuse_b)
  return __builtin_amdgcn_wmma_f32_16x16x32_bf16(false, a, false, b, (short)0, c,
                                                 false, false);
}

__device__ __forceinline__ v16bf combine8(v8bf lo, v8bf hi) {
  v16bf r;
#pragma unroll
  for (int i = 0; i < 8; ++i) { r[i] = lo[i]; r[i + 8] = hi[i]; }
  return r;
}

__device__ __forceinline__ void atomicAddF(float* p, float v) {
  unsafeAtomicAdd(p, v);   // global_atomic_add_f32 path on CDNA
}

// CDNA5 async global->LDS copy (ASYNCcnt path), ISA §15.18 VGLOBAL op 97.
// LDS byte offset = low 32 bits of the generic pointer (aperture truncation,
// ISA §10.2). 8 bytes per lane.
__device__ __forceinline__ void async_copy_b64(void* lds_dst, const void* gsrc) {
  uint32_t loff = (uint32_t)(uintptr_t)lds_dst;
  uint64_t gaddr = (uint64_t)(uintptr_t)gsrc;
  asm volatile("global_load_async_to_lds_b64 %0, %1, off"
               :: "v"(loff), "v"(gaddr) : "memory");
}
__device__ __forceinline__ void wait_asynccnt0() {
  asm volatile("s_wait_asynccnt 0x0" ::: "memory");
}

// ---------------- workspace layout (byte offsets, all 256B aligned) ---------
constexpr size_t SZ_XB      = (size_t)N * D * 2;      // bf16 X
constexpr size_t SZ_MUB     = (size_t)C * D * 2;      // bf16 mu / mu_newT
constexpr size_t OFF_XB     = 0;
constexpr size_t OFF_XBT    = OFF_XB    + SZ_XB;      // bf16 X^T [D,N]
constexpr size_t OFF_MUB    = OFF_XBT   + SZ_XB;      // bf16 mu  [C,D]
constexpr size_t OFF_MUNEWT = OFF_MUB   + SZ_MUB;     // bf16 mu_new^T [D,C]
constexpr size_t OFF_XNORM  = OFF_MUNEWT+ SZ_MUB;     // f32 [N]
constexpr size_t OFF_MUNORM = OFF_XNORM + (size_t)N * 4;  // f32 [C]
constexpr size_t OFF_SCOL   = OFF_MUNORM+ (size_t)C * 4;  // f32 [C]  (zeroed)
constexpr size_t OFF_INVS   = OFF_SCOL  + (size_t)C * 4;  // f32 [C]
constexpr size_t OFF_LOGPI  = OFF_INVS  + (size_t)C * 4;  // f32 [C]
constexpr size_t OFF_SCAL   = OFF_LOGPI + (size_t)C * 4;  // f32 [64] scalars
constexpr size_t OFF_MUACC  = OFF_SCAL  + 256;            // f32 [C,D] (zeroed)
constexpr size_t OFF_GAMMA  = OFF_MUACC + (size_t)C * D * 4; // bf16 [N,C]
// scalars: 0=sum g*D2, 1=sum g*lg, 2=sigma_new, 3=sum Sc*logpi,
//          4=sum Sc*||mu_new||^2, 5=sum ||Y||^2

// ---------------------------------------------------------------------------
__global__ __launch_bounds__(256) void k_zero(float* __restrict__ p, int n) {
  int i = blockIdx.x * blockDim.x + threadIdx.x;
  if (i < n) p[i] = 0.0f;
}

// One 128-thread block per row of X: bf16 copies (+transpose) and row norm.
__global__ __launch_bounds__(128)
void k_prep_x(const float* __restrict__ X, __bf16* __restrict__ Xb,
              __bf16* __restrict__ XbT, float* __restrict__ xnorm) {
  __shared__ float red[4];
  const int n = blockIdx.x, d = threadIdx.x;
  float x = X[(size_t)n * D + d];
  __bf16 b = (__bf16)x;
  Xb[(size_t)n * D + d] = b;
  XbT[(size_t)d * N + n] = b;
  float ss = x * x;
#pragma unroll
  for (int s = 1; s < 32; s <<= 1) ss += __shfl_xor(ss, s, 32);
  if ((threadIdx.x & 31) == 0) red[threadIdx.x >> 5] = ss;
  __syncthreads();
  if (threadIdx.x == 0) xnorm[n] = red[0] + red[1] + red[2] + red[3];
}

__global__ __launch_bounds__(128)
void k_prep_mu(const float* __restrict__ mu, __bf16* __restrict__ mub,
               float* __restrict__ munorm) {
  __shared__ float red[4];
  const int c = blockIdx.x, d = threadIdx.x;
  float x = mu[(size_t)c * D + d];
  mub[(size_t)c * D + d] = (__bf16)x;
  float ss = x * x;
#pragma unroll
  for (int s = 1; s < 32; s <<= 1) ss += __shfl_xor(ss, s, 32);
  if ((threadIdx.x & 31) == 0) red[threadIdx.x >> 5] = ss;
  __syncthreads();
  if (threadIdx.x == 0) munorm[c] = red[0] + red[1] + red[2] + red[3];
}

// ---------------------------------------------------------------------------
// K1: fused GEMM1 (X@mu^T via WMMA bf16) + row log-softmax over C.
// Workgroup = 8 waves, 16 N-rows; wave w owns columns [w*128, w*128+128).
__global__ __launch_bounds__(256)
void k_logits_softmax(const __bf16* __restrict__ Xb, const __bf16* __restrict__ mub,
                      const float* __restrict__ xnorm, const float* __restrict__ munorm,
                      const float* __restrict__ w, const float* __restrict__ sigma_p,
                      __bf16* __restrict__ gammaB, float* __restrict__ Scol,
                      float* __restrict__ scal) {
  __shared__ float ldsmax[16 * 8];
  __shared__ float ldssum[16 * 8];
  const int tid  = threadIdx.x;
  const int wave = tid >> 5;
  const int lane = tid & 31;
  const int m    = lane & 15;
  const int kh   = lane >> 4;       // which 16-lane half
  const int row0 = blockIdx.x * 16;
  const int colw = wave * 128;

  const float sig    = sigma_p[0];
  const float twos2  = 2.0f * sig * sig;
  const float inv2s2 = 1.0f / twos2;

  v8f acc[8];
#pragma unroll
  for (int t = 0; t < 8; ++t) acc[t] = (v8f){0,0,0,0,0,0,0,0};

#pragma unroll
  for (int kb = 0; kb < 4; ++kb) {            // K = 128 in 4 blocks of 32
    const __bf16* arow = Xb + (size_t)(row0 + m) * D + kb * 32;
    v16bf a = combine8(*(const v8bf*)(arow + kh * 8),
                       *(const v8bf*)(arow + 16 + kh * 8));
#pragma unroll
    for (int t = 0; t < 8; ++t) {
      const int col = colw + t * 16 + m;      // B column = mu row
      v16bf b = combine8(*(const v8bf*)(mub + (size_t)col * D + kb * 32 + kh * 8),
                         *(const v8bf*)(mub + (size_t)col * D + kb * 32 + 16 + kh * 8));
      acc[t] = wmma_bf16(a, b, acc[t]);
    }
  }

  // logits = w_c - D2/(2 sigma^2);  D2 = |x|^2 - 2 dot + |mu|^2
  float xn[8];
#pragma unroll
  for (int j = 0; j < 8; ++j) xn[j] = xnorm[row0 + j + 8 * kh];
  float wv[8];
#pragma unroll
  for (int t = 0; t < 8; ++t) {
    const int col = colw + t * 16 + m;
    const float wc = w[col], mn = munorm[col];
    wv[t] = wc;
#pragma unroll
    for (int j = 0; j < 8; ++j) {
      float d2 = xn[j] - 2.0f * acc[t][j] + mn;
      acc[t][j] = wc - d2 * inv2s2;           // overwrite with logit
    }
  }

  // row max over full C (intra-wave shfl + cross-wave LDS)
  float rmax[8];
#pragma unroll
  for (int j = 0; j < 8; ++j) {
    float v = -3.0e38f;
#pragma unroll
    for (int t = 0; t < 8; ++t) v = fmaxf(v, acc[t][j]);
#pragma unroll
    for (int s = 1; s < 16; s <<= 1) v = fmaxf(v, __shfl_xor(v, s, 32));
    rmax[j] = v;
    if (m == 0) ldsmax[(j + 8 * kh) * 8 + wave] = v;
  }
  __syncthreads();
#pragma unroll
  for (int j = 0; j < 8; ++j) {
    float v = rmax[j];
#pragma unroll
    for (int w2 = 0; w2 < 8; ++w2) v = fmaxf(v, ldsmax[(j + 8 * kh) * 8 + w2]);
    rmax[j] = v;
  }
  // row sum of exp
  float lls[8];
#pragma unroll
  for (int j = 0; j < 8; ++j) {
    float v = 0.0f;
#pragma unroll
    for (int t = 0; t < 8; ++t) v += expf(acc[t][j] - rmax[j]);
#pragma unroll
    for (int s = 1; s < 16; s <<= 1) v += __shfl_xor(v, s, 32);
    if (m == 0) ldssum[(j + 8 * kh) * 8 + wave] = v;
  }
  __syncthreads();
#pragma unroll
  for (int j = 0; j < 8; ++j) {
    float v = 0.0f;
#pragma unroll
    for (int w2 = 0; w2 < 8; ++w2) v += ldssum[(j + 8 * kh) * 8 + w2];
    lls[j] = logf(v);
  }

  // gamma, stores, column sums, scalar accumulators
  float lgd2 = 0.0f, lgl = 0.0f;
#pragma unroll
  for (int t = 0; t < 8; ++t) {
    const int col = colw + t * 16 + m;
    float cs = 0.0f;
#pragma unroll
    for (int j = 0; j < 8; ++j) {
      const float logit = acc[t][j];
      const float lg = logit - rmax[j] - lls[j];
      const float g  = expf(lg);
      const int  row = row0 + j + 8 * kh;
      gammaB[(size_t)row * C + col] = (__bf16)g;
      const float d2 = (wv[t] - logit) * twos2;
      lgd2 += g * d2;
      lgl  += g * lg;
      cs   += g;
    }
    cs += __shfl_xor(cs, 16, 32);
    if (kh == 0) atomicAddF(&Scol[col], cs);
  }
#pragma unroll
  for (int s = 1; s < 32; s <<= 1) {
    lgd2 += __shfl_xor(lgd2, s, 32);
    lgl  += __shfl_xor(lgl,  s, 32);
  }
  if (lane == 0) { atomicAddF(&scal[0], lgd2); atomicAddF(&scal[1], lgl); }
}

// ---------------------------------------------------------------------------
// small block-wide reductions (blockDim.x == 1024)
__device__ __forceinline__ float block_sum(float v, float* lds) {
  const int lane = threadIdx.x & 31, wid = threadIdx.x >> 5;
#pragma unroll
  for (int s = 1; s < 32; s <<= 1) v += __shfl_xor(v, s, 32);
  __syncthreads();
  if (lane == 0) lds[wid] = v;
  __syncthreads();
  if (wid == 0) {
    float t = lds[lane];
#pragma unroll
    for (int s = 1; s < 32; s <<= 1) t += __shfl_xor(t, s, 32);
    if (lane == 0) lds[0] = t;
  }
  __syncthreads();
  return lds[0];
}
__device__ __forceinline__ float block_max(float v, float* lds) {
  const int lane = threadIdx.x & 31, wid = threadIdx.x >> 5;
#pragma unroll
  for (int s = 1; s < 32; s <<= 1) v = fmaxf(v, __shfl_xor(v, s, 32));
  __syncthreads();
  if (lane == 0) lds[wid] = v;
  __syncthreads();
  if (wid == 0) {
    float t = lds[lane];
#pragma unroll
    for (int s = 1; s < 32; s <<= 1) t = fmaxf(t, __shfl_xor(t, s, 32));
    if (lane == 0) lds[0] = t;
  }
  __syncthreads();
  return lds[0];
}

// h_c = log S_c, w_new, logpi, sigma_new, sum Sc*logpi
__global__ __launch_bounds__(1024)
void k_stats(const float* __restrict__ Scol, float* __restrict__ invS,
             float* __restrict__ logpi, float* __restrict__ scal) {
  __shared__ float lds[32];
  const int c = threadIdx.x;
  const float Sc = fmaxf(Scol[c], 1e-30f);
  const float h  = logf(Sc);
  const float hm = block_sum(h, lds) * (1.0f / (float)C);
  const float wn = h - hm;
  const float mx = block_max(wn, lds);
  const float se = block_sum(expf(wn - mx), lds);
  const float lp = wn - (mx + logf(se));
  logpi[c] = lp;
  invS[c]  = 1.0f / Sc;
  const float slp = block_sum(Sc * lp, lds);
  if (c == 0) {
    scal[3] = slp;
    scal[2] = sqrtf(scal[0] / ((float)D * (float)N));   // sigma_new
  }
}

// ---------------------------------------------------------------------------
// K2: mu accumulator = gamma^T @ X (WMMA bf16, K split over N, f32 atomics).
// grid (C/32, 16); block = 8 waves; wave w owns d-block w*16 and TWO c-tiles
// (independent WMMA chains sharing one B fragment).
// gamma tile staged into LDS with CDNA5 async global->LDS DMA (ASYNCcnt),
// transposed at read time (A-fragment gather per the ISA VGPR layout).
__global__ __launch_bounds__(256)
void k_mu_gemm(const __bf16* __restrict__ gammaB, const __bf16* __restrict__ XbT,
               float* __restrict__ muacc) {
  __shared__ __align__(16) __bf16 raw[32 * 32];     // [n-local][c-local] 2KB
  const int tid = threadIdx.x;
  const int wave = tid >> 5, lane = tid & 31, m = lane & 15, kh = lane >> 4;
  const int cb = blockIdx.x * 32;
  const int n0 = blockIdx.y * (N / 16);             // 2048-row K chunk
  const int db = wave * 16;
  v8f acc0 = (v8f){0,0,0,0,0,0,0,0};
  v8f acc1 = (v8f){0,0,0,0,0,0,0,0};

  const int sr = tid >> 3;            // 0..31  (staging row n)
  const int sc = (tid & 7) * 4;       // 0..28  (staging col c, 4 bf16 = 8B)

  for (int kb = 0; kb < (N / 16) / 32; ++kb) {      // 64 K-blocks of 32
    const int nb = n0 + kb * 32;
    __syncthreads();                                 // LDS reuse fence
    async_copy_b64(&raw[sr * 32 + sc],
                   gammaB + (size_t)(nb + sr) * C + cb + sc);
    wait_asynccnt0();
    __syncthreads();

    // A fragments (gamma^T): rows = c-local, K = n-local (read-time transpose)
    v16bf a0, a1;
#pragma unroll
    for (int e = 0; e < 8; ++e) {
      const int k0 = kh * 8 + e;
      const int k1 = 16 + kh * 8 + e;
      a0[e]     = raw[k0 * 32 + m];
      a0[e + 8] = raw[k1 * 32 + m];
      a1[e]     = raw[k0 * 32 + 16 + m];
      a1[e + 8] = raw[k1 * 32 + 16 + m];
    }
    v16bf b = combine8(*(const v8bf*)(XbT + (size_t)(db + m) * N + nb + kh * 8),
                       *(const v8bf*)(XbT + (size_t)(db + m) * N + nb + 16 + kh * 8));
    acc0 = wmma_bf16(a0, b, acc0);
    acc1 = wmma_bf16(a1, b, acc1);
  }
#pragma unroll
  for (int j = 0; j < 8; ++j) {
    const int d = db + m;
    atomicAddF(&muacc[(cb + j + 8 * kh) * D + d],      acc0[j]);
    atomicAddF(&muacc[(cb + 16 + j + 8 * kh) * D + d], acc1[j]);
  }
}

// mu_new = muacc / S_c; write transposed bf16; accumulate Sc*||mu_new||^2
__global__ __launch_bounds__(128)
void k_mu_finalize(const float* __restrict__ muacc, const float* __restrict__ invS,
                   const float* __restrict__ Scol, __bf16* __restrict__ munewT,
                   float* __restrict__ scal) {
  __shared__ float red[4];
  const int c = blockIdx.x, d = threadIdx.x;
  const float v = muacc[c * D + d] * invS[c];
  munewT[(size_t)d * C + c] = (__bf16)v;
  float ss = v * v;
#pragma unroll
  for (int s = 1; s < 32; s <<= 1) ss += __shfl_xor(ss, s, 32);
  if ((threadIdx.x & 31) == 0) red[threadIdx.x >> 5] = ss;
  __syncthreads();
  if (threadIdx.x == 0)
    atomicAddF(&scal[4], Scol[c] * (red[0] + red[1] + red[2] + red[3]));
}

// ---------------------------------------------------------------------------
// K3: Y = gamma @ mu_new (WMMA bf16) + sum ||Y||^2.
// Workgroup = 32 N-rows; wave owns d-block w*16 and TWO n-tiles sharing B.
__global__ __launch_bounds__(256)
void k_y_gemm(const __bf16* __restrict__ gammaB, const __bf16* __restrict__ munewT,
              float* __restrict__ Y, float* __restrict__ scal) {
  const int tid = threadIdx.x;
  const int wave = tid >> 5, lane = tid & 31, m = lane & 15, kh = lane >> 4;
  const int n0 = blockIdx.x * 32;
  const int db = wave * 16;
  v8f acc0 = (v8f){0,0,0,0,0,0,0,0};
  v8f acc1 = (v8f){0,0,0,0,0,0,0,0};
  for (int kb = 0; kb < C / 32; ++kb) {
    const int k0 = kb * 32;
    const __bf16* ar0 = gammaB + (size_t)(n0 + m) * C + k0;
    const __bf16* ar1 = gammaB + (size_t)(n0 + 16 + m) * C + k0;
    v16bf a0 = combine8(*(const v8bf*)(ar0 + kh * 8),
                        *(const v8bf*)(ar0 + 16 + kh * 8));
    v16bf a1 = combine8(*(const v8bf*)(ar1 + kh * 8),
                        *(const v8bf*)(ar1 + 16 + kh * 8));
    v16bf b = combine8(*(const v8bf*)(munewT + (size_t)(db + m) * C + k0 + kh * 8),
                       *(const v8bf*)(munewT + (size_t)(db + m) * C + k0 + 16 + kh * 8));
    acc0 = wmma_bf16(a0, b, acc0);
    acc1 = wmma_bf16(a1, b, acc1);
  }
  float loc = 0.0f;
#pragma unroll
  for (int j = 0; j < 8; ++j) {
    const int d = db + m;
    const int na = n0 + j + 8 * kh;
    const int nb2 = n0 + 16 + j + 8 * kh;
    const float y0 = acc0[j], y1 = acc1[j];
    Y[(size_t)na * D + d]  = y0;
    Y[(size_t)nb2 * D + d] = y1;
    loc += y0 * y0 + y1 * y1;
  }
#pragma unroll
  for (int s = 1; s < 32; s <<= 1) loc += __shfl_xor(loc, s, 32);
  if (lane == 0) atomicAddF(&scal[5], loc);
}

// Cfe = (sum Sc*||mu||^2 - sum||Y||^2)/(2 s^2) + N*D*log(s) - sum Sc*logpi + sum g*lg
__global__ void k_final(const float* __restrict__ scal, float* __restrict__ out) {
  const float sn = scal[2];
  const float cfe = (scal[4] - scal[5]) / (2.0f * sn * sn)
                  + (float)N * (float)D * logf(sn)
                  - scal[3] + scal[1];
  out[(size_t)N * D] = cfe;
}

// ---------------------------------------------------------------------------
extern "C" void kernel_launch(void* const* d_in, const int* in_sizes, int n_in,
                              void* d_out, int out_size, void* d_ws, size_t ws_size,
                              hipStream_t stream) {
  (void)in_sizes; (void)n_in; (void)out_size; (void)ws_size;
  const float* X     = (const float*)d_in[0];
  const float* mu    = (const float*)d_in[1];
  const float* w     = (const float*)d_in[2];
  const float* sigma = (const float*)d_in[3];
  float* out = (float*)d_out;
  char*  ws  = (char*)d_ws;

  __bf16* Xb     = (__bf16*)(ws + OFF_XB);
  __bf16* XbT    = (__bf16*)(ws + OFF_XBT);
  __bf16* mub    = (__bf16*)(ws + OFF_MUB);
  __bf16* munewT = (__bf16*)(ws + OFF_MUNEWT);
  float*  xnorm  = (float*)(ws + OFF_XNORM);
  float*  munorm = (float*)(ws + OFF_MUNORM);
  float*  Scol   = (float*)(ws + OFF_SCOL);
  float*  invS   = (float*)(ws + OFF_INVS);
  float*  logpi  = (float*)(ws + OFF_LOGPI);
  float*  scal   = (float*)(ws + OFF_SCAL);
  float*  muacc  = (float*)(ws + OFF_MUACC);
  __bf16* gammaB = (__bf16*)(ws + OFF_GAMMA);

  // zero the accumulator region [Scol .. muacc end)
  const int nz = (int)((OFF_GAMMA - OFF_SCOL) / 4);
  k_zero<<<(nz + 255) / 256, 256, 0, stream>>>(Scol, nz);

  k_prep_x<<<N, 128, 0, stream>>>(X, Xb, XbT, xnorm);
  k_prep_mu<<<C, 128, 0, stream>>>(mu, mub, munorm);

  k_logits_softmax<<<N / 16, 256, 0, stream>>>(Xb, mub, xnorm, munorm, w, sigma,
                                               gammaB, Scol, scal);

  k_stats<<<1, 1024, 0, stream>>>(Scol, invS, logpi, scal);

  k_mu_gemm<<<dim3(C / 32, 16), 256, 0, stream>>>(gammaB, XbT, muacc);
  k_mu_finalize<<<C, 128, 0, stream>>>(muacc, invS, Scol, munewT, scal);

  k_y_gemm<<<N / 32, 256, 0, stream>>>(gammaB, munewT, out, scal);
  k_final<<<1, 1, 0, stream>>>(scal, out);
}
